// CustomBertSelfAttention_49314814492808
// MI455X (gfx1250) — matrix operand order repaired and checked by reading, not compiled, not run on hardware
//
#include <hip/hip_runtime.h>
#include <stdint.h>

// ---------------- CDNA5 WMMA types ----------------
typedef __attribute__((ext_vector_type(16))) __bf16 v16bf;
typedef __attribute__((ext_vector_type(8)))  float  v8f;
static_assert(sizeof(v16bf) == 32, "v16bf");
static_assert(sizeof(v8f) == 32, "v8f");

#define LDT 40  // LDS row stride (ushorts) for 32-wide K tiles; 80B = 5*16B aligned

#define AS1 __attribute__((address_space(1)))
#define AS3 __attribute__((address_space(3)))

// Builtin's pointee type per the compiler diagnostic: int __vector(4).
typedef int vsi4 __attribute__((vector_size(16)));
#define GLOBAL_V4(p) ((AS1 vsi4*)(p))
#define LDS_V4(p)    ((AS3 vsi4*)(p))

// Async global->LDS DMA (gfx1250, tracked by ASYNCcnt) if the toolchain has it.
#if defined(__has_builtin)
#if __has_builtin(__builtin_amdgcn_global_load_async_to_lds_b128)
#define HAVE_ASYNC_COPY 1
#endif
#endif
#ifndef HAVE_ASYNC_COPY
#define HAVE_ASYNC_COPY 0
#endif

__device__ inline void wait_async0() {
#if HAVE_ASYNC_COPY
#if defined(__has_builtin) && __has_builtin(__builtin_amdgcn_s_wait_asynccnt)
  __builtin_amdgcn_s_wait_asynccnt(0);
#else
  asm volatile("s_wait_asynccnt 0x0" ::: "memory");
#endif
#endif
}

union FragU {
  v16bf v;
  uint4 q[2];
  unsigned short u[16];
};

__device__ inline unsigned short f32_to_bf16(float f) {
  union { float f; uint32_t u; } c;
  c.f = f;
  uint32_t u = c.u;
  u += 0x7FFFu + ((u >> 16) & 1u);  // round-to-nearest-even
  return (unsigned short)(u >> 16);
}

// ---- Fragment loads matching ISA 7.12.2 VGPR layouts (wave32) ----
// A 16x32 bf16: lane<16 row M=l holds K=0..7 (V0..3) and K=16..23 (V4..7);
//               lane>=16 holds K=8..15 and K=24..31.
__device__ inline v16bf load_fragA(const unsigned short* rowp, int half) {
  FragU f;
  f.q[0] = *(const uint4*)(rowp + half * 8);
  f.q[1] = *(const uint4*)(rowp + 16 + half * 8);
  return f.v;
}
// B 32x16 bf16 (stored K-major per column n): lane<16 col N=l holds K=0..7 and K=16..23,
// lane>=16 holds K=8..15 and K=24..31 (same order as A).
__device__ inline v16bf load_fragB(const unsigned short* rowp, int half) {
  FragU f;
  f.q[0] = *(const uint4*)(rowp + half * 8);
  f.q[1] = *(const uint4*)(rowp + 16 + half * 8);
  return f.v;
}

// One K=32 step: wave computes 64(M) x 32(N) via 8 wmma ops.
__device__ inline void wmma_step(const unsigned short* Ald, const unsigned short* Btld,
                                 v8f acc[4][2], int waveM, int waveN, int lane) {
  const int half = lane >> 4, l = lane & 15;
  v16bf a[4], b[2];
#pragma unroll
  for (int mt = 0; mt < 4; ++mt)
    a[mt] = load_fragA(Ald + (size_t)(waveM * 64 + mt * 16 + l) * LDT, half);
#pragma unroll
  for (int nt = 0; nt < 2; ++nt)
    b[nt] = load_fragB(Btld + (size_t)(waveN * 32 + nt * 16 + l) * LDT, half);
#pragma unroll
  for (int mt = 0; mt < 4; ++mt)
#pragma unroll
    for (int nt = 0; nt < 2; ++nt)
      acc[mt][nt] = __builtin_amdgcn_wmma_f32_16x16x32_bf16(
          false, a[mt], false, b[nt], (short)0, acc[mt][nt], false, false);
}

// Cooperative tile loads (256 threads).
// A tile: 128 rows x 32 K from row-major bf16 src (ld elems/row).
// Uses async global->LDS DMA (no VGPR transit) when available.
__device__ inline void load_tileA(unsigned short* Ald, const unsigned short* src,
                                  size_t row0, int k0, int ld, int tid) {
  const int r = tid >> 1;
  const int c = (tid & 1) * 16;
  const unsigned short* g = src + (row0 + (size_t)r) * (size_t)ld + k0 + c;
  unsigned short* l = Ald + (size_t)r * LDT + c;
#if HAVE_ASYNC_COPY
  __builtin_amdgcn_global_load_async_to_lds_b128(GLOBAL_V4(g), LDS_V4(l), 0, 0);
  __builtin_amdgcn_global_load_async_to_lds_b128(GLOBAL_V4(g + 8), LDS_V4(l + 8), 0, 0);
#else
  uint4 q0 = *(const uint4*)(g);
  uint4 q1 = *(const uint4*)(g + 8);
  *(uint4*)(l) = q0;
  *(uint4*)(l + 8) = q1;
#endif
}

// Bt tile from an N-major source (B[k][n], n contiguous): transpose into Bt[n][k].
// Each thread handles an 8(K) x 2(N) sliver: 8 coalesced dword loads, register
// transpose, then TWO ds_store_b128 (full 16B K-runs of one column each).
__device__ inline void load_tileBt_transpose(unsigned short* Btld, const unsigned short* src,
                                             size_t krow0, int n0, int ld, int tid) {
  const int ko = (tid >> 6) * 8;   // 0,8,16,24
  const int n  = (tid & 63) * 2;   // 0..126
  const unsigned short* g = src + (krow0 + (size_t)ko) * (size_t)ld + n0 + n;
  uint32_t w[8];
#pragma unroll
  for (int j = 0; j < 8; ++j) w[j] = *(const uint32_t*)(g + (size_t)j * ld);
  // w[j] = { col n (lo16), col n+1 (hi16) } at k = ko + j.
  uint4 c0, c1;
  {
    uint32_t* p0 = (uint32_t*)&c0;
    uint32_t* p1 = (uint32_t*)&c1;
#pragma unroll
    for (int j = 0; j < 4; ++j) {
      const uint32_t lo0 = w[2 * j] & 0xFFFFu, hi0 = w[2 * j] >> 16;
      const uint32_t lo1 = w[2 * j + 1] & 0xFFFFu, hi1 = w[2 * j + 1] >> 16;
      p0[j] = lo0 | (lo1 << 16);  // column n:   k = ko+2j, ko+2j+1
      p1[j] = hi0 | (hi1 << 16);  // column n+1
    }
  }
  *(uint4*)(Btld + (size_t)n * LDT + ko)       = c0;  // 16B aligned: 80B rows, ko*2 in {0,16,32,48}
  *(uint4*)(Btld + (size_t)(n + 1) * LDT + ko) = c1;
}

// ---------------- Kernel 1: fp32 -> bf16 cast ----------------
__global__ void cvt_bf16_kernel(const float* __restrict__ src,
                                unsigned short* __restrict__ dst, size_t n) {
  size_t i = (size_t)blockIdx.x * blockDim.x + threadIdx.x;
  const size_t stride = (size_t)gridDim.x * blockDim.x;
  for (; i < n; i += stride) dst[i] = f32_to_bf16(src[i]);
}

// ---------------- Kernel 2: fused QKV GEMM (bf16 WMMA, f32 acc) ----------------
// z selects Wq/Wk/Wv slab & bias; out bf16 [16384 x 1024] per slab.
__global__ __launch_bounds__(256) void qkv_gemm_kernel(
    const unsigned short* __restrict__ Hbf, const unsigned short* __restrict__ Wbf,
    const float* __restrict__ bq, const float* __restrict__ bk,
    const float* __restrict__ bv, unsigned short* __restrict__ QKVbf) {
  __shared__ alignas(16) unsigned short Ald[2][128 * LDT];
  __shared__ alignas(16) unsigned short Btld[2][128 * LDT];
  const int tid = threadIdx.x, lane = tid & 31, wave = tid >> 5;
  const int waveM = wave >> 2, waveN = wave & 3;
  const int z = blockIdx.z;
  const size_t mTile = (size_t)blockIdx.y * 128;
  const int nTile = blockIdx.x * 128;
  const unsigned short* Wz = Wbf + (size_t)z * 1024 * 1024;
  const float* bias = (z == 0) ? bq : ((z == 1) ? bk : bv);
  unsigned short* Out = QKVbf + (size_t)z * 16384 * 1024;

  v8f zero = {};
  v8f acc[4][2];
#pragma unroll
  for (int mt = 0; mt < 4; ++mt)
#pragma unroll
    for (int nt = 0; nt < 2; ++nt) acc[mt][nt] = zero;

  load_tileA(Ald[0], Hbf, mTile, 0, 1024, tid);
  load_tileBt_transpose(Btld[0], Wz, 0, nTile, 1024, tid);
  const int NSTEP = 1024 / 32;
  for (int step = 0; step < NSTEP; ++step) {
    wait_async0();      // my async fills (prev iter / preload) complete
    __syncthreads();    // everyone's fills visible
    const int cur = step & 1, nxt = cur ^ 1;
    if (step + 1 < NSTEP) {
      load_tileA(Ald[nxt], Hbf, mTile, (step + 1) * 32, 1024, tid);
      load_tileBt_transpose(Btld[nxt], Wz, (size_t)(step + 1) * 32, nTile, 1024, tid);
    }
    wmma_step(Ald[cur], Btld[cur], acc, waveM, waveN, lane);
  }

  const int half = lane >> 4, l = lane & 15;
#pragma unroll
  for (int mt = 0; mt < 4; ++mt)
#pragma unroll
    for (int nt = 0; nt < 2; ++nt) {
      const int col = nTile + waveN * 32 + nt * 16 + l;
      const float bcol = bias[col];
      const size_t rbase = mTile + waveM * 64 + mt * 16 + half * 8;
#pragma unroll
      for (int r = 0; r < 8; ++r)
        Out[(rbase + r) * 1024 + col] = f32_to_bf16(acc[mt][nt][r] + bcol);
    }
}

// ---------------- Kernel 3: scores = Q @ K^T * 0.125 + mask (fp32 out) -------
__global__ __launch_bounds__(256) void scores_gemm_kernel(
    const unsigned short* __restrict__ Qbf, const unsigned short* __restrict__ Kbf,
    const float* __restrict__ mask, float* __restrict__ S) {
  __shared__ alignas(16) unsigned short Ald[2][128 * LDT];
  __shared__ alignas(16) unsigned short Btld[2][128 * LDT];
  const int tid = threadIdx.x, lane = tid & 31, wave = tid >> 5;
  const int waveM = wave >> 2, waveN = wave & 3;
  const int b = blockIdx.z;
  const size_t rowbase = (size_t)b * 2048;
  const size_t sTile = (size_t)blockIdx.y * 128;  // query rows
  const int tTile = blockIdx.x * 128;             // key cols

  v8f zero = {};
  v8f acc[4][2];
#pragma unroll
  for (int mt = 0; mt < 4; ++mt)
#pragma unroll
    for (int nt = 0; nt < 2; ++nt) acc[mt][nt] = zero;

  load_tileA(Ald[0], Qbf, rowbase + sTile, 0, 1024, tid);
  // B = K^T, so Bt[n][k] = K[n][k]: K rows load straight in, no transpose.
  load_tileA(Btld[0], Kbf, rowbase + (size_t)tTile, 0, 1024, tid);
  const int NSTEP = 1024 / 32;
  for (int step = 0; step < NSTEP; ++step) {
    wait_async0();
    __syncthreads();
    const int cur = step & 1, nxt = cur ^ 1;
    if (step + 1 < NSTEP) {
      load_tileA(Ald[nxt], Qbf, rowbase + sTile, (step + 1) * 32, 1024, tid);
      load_tileA(Btld[nxt], Kbf, rowbase + (size_t)tTile, (step + 1) * 32, 1024, tid);
    }
    wmma_step(Ald[cur], Btld[cur], acc, waveM, waveN, lane);
  }

  const int half = lane >> 4, l = lane & 15;
#pragma unroll
  for (int mt = 0; mt < 4; ++mt)
#pragma unroll
    for (int nt = 0; nt < 2; ++nt) {
      const int t = tTile + waveN * 32 + nt * 16 + l;
      const float m = mask[(size_t)b * 2048 + t];
      const size_t srow = sTile + waveM * 64 + mt * 16 + half * 8;
#pragma unroll
      for (int r = 0; r < 8; ++r)
        S[((size_t)b * 2048 + srow + r) * 2048 + t] = acc[mt][nt][r] * 0.125f + m;
    }
}

// ---------------- Kernel 4: row softmax fp32 -> bf16 P ----------------
__global__ __launch_bounds__(256) void softmax_kernel(const float* __restrict__ S,
                                                      unsigned short* __restrict__ Pbf) {
  const int s = blockIdx.x, b = blockIdx.y, tid = threadIdx.x;
  const float* row = S + ((size_t)b * 2048 + s) * 2048;
  unsigned short* prow = Pbf + ((size_t)b * 2048 + s) * 2048;
  __shared__ float red[256];

  float v[8];
#pragma unroll
  for (int i = 0; i < 8; ++i) v[i] = row[tid * 8 + i];

  float m = v[0];
#pragma unroll
  for (int i = 1; i < 8; ++i) m = fmaxf(m, v[i]);
  red[tid] = m;
  __syncthreads();
  for (int off = 128; off > 0; off >>= 1) {
    if (tid < off) red[tid] = fmaxf(red[tid], red[tid + off]);
    __syncthreads();
  }
  m = red[0];
  __syncthreads();

  float sum = 0.f;
#pragma unroll
  for (int i = 0; i < 8; ++i) {
    v[i] = __expf(v[i] - m);
    sum += v[i];
  }
  red[tid] = sum;
  __syncthreads();
  for (int off = 128; off > 0; off >>= 1) {
    if (tid < off) red[tid] += red[tid + off];
    __syncthreads();
  }
  const float inv = 1.0f / red[0];
#pragma unroll
  for (int i = 0; i < 8; ++i) prow[tid * 8 + i] = f32_to_bf16(v[i] * inv);
}

// ---------------- Kernel 5: context = P @ V (fp32 out) ----------------
__global__ __launch_bounds__(256) void ctx_gemm_kernel(
    const unsigned short* __restrict__ Pbf, const unsigned short* __restrict__ Vbf,
    float* __restrict__ Out) {
  __shared__ alignas(16) unsigned short Ald[2][128 * LDT];
  __shared__ alignas(16) unsigned short Btld[2][128 * LDT];
  const int tid = threadIdx.x, lane = tid & 31, wave = tid >> 5;
  const int waveM = wave >> 2, waveN = wave & 3;
  const int b = blockIdx.z;
  const size_t sTile = (size_t)blockIdx.y * 128;
  const int nTile = blockIdx.x * 128;
  const unsigned short* Pb = Pbf + (size_t)b * 2048 * 2048;

  v8f zero = {};
  v8f acc[4][2];
#pragma unroll
  for (int mt = 0; mt < 4; ++mt)
#pragma unroll
    for (int nt = 0; nt < 2; ++nt) acc[mt][nt] = zero;

  load_tileA(Ald[0], Pb, sTile, 0, 2048, tid);
  load_tileBt_transpose(Btld[0], Vbf, (size_t)b * 2048, nTile, 1024, tid);
  const int NSTEP = 2048 / 32;  // K = keys dimension
  for (int step = 0; step < NSTEP; ++step) {
    wait_async0();
    __syncthreads();
    const int cur = step & 1, nxt = cur ^ 1;
    if (step + 1 < NSTEP) {
      load_tileA(Ald[nxt], Pb, sTile, (step + 1) * 32, 2048, tid);
      load_tileBt_transpose(Btld[nxt], Vbf, (size_t)b * 2048 + (step + 1) * 32, nTile, 1024, tid);
    }
    wmma_step(Ald[cur], Btld[cur], acc, waveM, waveN, lane);
  }

  const int half = lane >> 4, l = lane & 15;
#pragma unroll
  for (int mt = 0; mt < 4; ++mt)
#pragma unroll
    for (int nt = 0; nt < 2; ++nt) {
      const int d = nTile + waveN * 32 + nt * 16 + l;
      const size_t srow = sTile + waveM * 64 + mt * 16 + half * 8;
#pragma unroll
      for (int r = 0; r < 8; ++r)
        Out[((size_t)b * 2048 + srow + r) * 1024 + d] = acc[mt][nt][r];
    }
}

// ---------------- Host orchestration ----------------
extern "C" void kernel_launch(void* const* d_in, const int* in_sizes, int n_in,
                              void* d_out, int out_size, void* d_ws, size_t ws_size,
                              hipStream_t stream) {
  const float* H    = (const float*)d_in[0];
  const float* mask = (const float*)d_in[1];
  const float* Wq   = (const float*)d_in[2];
  const float* bq   = (const float*)d_in[3];
  const float* Wk   = (const float*)d_in[4];
  const float* bk   = (const float*)d_in[5];
  const float* Wv   = (const float*)d_in[6];
  const float* bv   = (const float*)d_in[7];
  float* out = (float*)d_out;

  // Workspace layout (bytes), all offsets 16B-aligned. Total ~326 MB.
  char* ws = (char*)d_ws;
  unsigned short* Hbf = (unsigned short*)ws;            ws += 16384ULL * 1024 * 2;    // 32 MB
  unsigned short* Wbf = (unsigned short*)ws;            ws += 3ULL * 1024 * 1024 * 2; // 6 MB
  unsigned short* Qbf = (unsigned short*)ws;            ws += 16384ULL * 1024 * 2;    // 32 MB
  unsigned short* Kbf = (unsigned short*)ws;            ws += 16384ULL * 1024 * 2;    // 32 MB
  unsigned short* Vbf = (unsigned short*)ws;            ws += 16384ULL * 1024 * 2;    // 32 MB
  float*          Sf  = (float*)ws;                     ws += 8ULL * 2048 * 2048 * 4; // 128 MB
  unsigned short* Pbf = (unsigned short*)ws;            ws += 8ULL * 2048 * 2048 * 2; // 64 MB
  (void)ws_size; (void)in_sizes; (void)n_in; (void)out_size;

  // 1) casts to bf16
  cvt_bf16_kernel<<<4096, 256, 0, stream>>>(H, Hbf, 16384ULL * 1024);
  cvt_bf16_kernel<<<1024, 256, 0, stream>>>(Wq, Wbf + 0ULL * 1024 * 1024, 1024ULL * 1024);
  cvt_bf16_kernel<<<1024, 256, 0, stream>>>(Wk, Wbf + 1ULL * 1024 * 1024, 1024ULL * 1024);
  cvt_bf16_kernel<<<1024, 256, 0, stream>>>(Wv, Wbf + 2ULL * 1024 * 1024, 1024ULL * 1024);

  // 2) Q/K/V = H @ W + b   (Qbf/Kbf/Vbf are contiguous slabs after Qbf)
  dim3 gq(1024 / 128, 16384 / 128, 3);
  qkv_gemm_kernel<<<gq, 256, 0, stream>>>(Hbf, Wbf, bq, bk, bv, Qbf);

  // 3) scores = Q @ K^T / sqrt(64) + mask
  dim3 gs(2048 / 128, 2048 / 128, 8);
  scores_gemm_kernel<<<gs, 256, 0, stream>>>(Qbf, Kbf, mask, Sf);

  // 4) softmax rows -> bf16 probabilities (scores stay resident in 192MB L2)
  dim3 gsm(2048, 8);
  softmax_kernel<<<gsm, 256, 0, stream>>>(Sf, Pbf);

  // 5) context = P @ V
  dim3 gc(1024 / 128, 2048 / 128, 8);
  ctx_gemm_kernel<<<gc, 256, 0, stream>>>(Pbf, Vbf, out);
}
